// LBLHighwayBiLmV2_20160576487472
// MI455X (gfx1250) — matrix operand order repaired and checked by reading, not compiled, not run on hardware
//
#include <hip/hip_runtime.h>
#include <hip/hip_bf16.h>

typedef __attribute__((ext_vector_type(16))) __bf16 v16bf;
typedef __attribute__((ext_vector_type(8)))  __bf16 v8bf;
typedef __attribute__((ext_vector_type(8)))  float  v8f;

#define BB 16
#define SS 2048
#define DD 512
#define MM (BB * SS)          // 32768 rows
#define NPROJ 1024            // 2*D

// ---------------------------------------------------------------------------
// Weight convert + transpose: W[k][n] f32 -> Wt[n][k] bf16, 8 matrices
// (dir 0/1 x layer 0/1 x highway 0/1), matrix index z = dir*4 + l*2 + h.
// ---------------------------------------------------------------------------
__global__ __launch_bounds__(256) void wt_transpose_kernel(
    const float* __restrict__ fw_W, const float* __restrict__ bw_W,
    __bf16* __restrict__ Wt)
{
    __shared__ float tile[32][33];
    const int z = blockIdx.z;
    const float* src = (z & 4) ? bw_W : fw_W;
    src += (size_t)(z & 3) * DD * NPROJ;
    __bf16* dst = Wt + (size_t)z * NPROJ * DD;

    const int n0 = blockIdx.x * 32;   // column of W / row of Wt, [0,1024)
    const int k0 = blockIdx.y * 32;   // row of W / column of Wt, [0,512)
    const int tx = threadIdx.x & 31;
    const int ty = threadIdx.x >> 5;  // 8 rows of 32

    #pragma unroll
    for (int i = ty; i < 32; i += 8)
        tile[i][tx] = src[(size_t)(k0 + i) * NPROJ + n0 + tx];
    __syncthreads();
    #pragma unroll
    for (int i = ty; i < 32; i += 8)
        dst[(size_t)(n0 + i) * DD + k0 + tx] = (__bf16)tile[tx][i];
}

// ---------------------------------------------------------------------------
// Banded conv along S, both directions. blockIdx: x = t, y = b, z = dir.
// dir 0 (causal):      out[t] = sum_{d=0..16} w[16-d] * x[t-d]
// dir 1 (anticausal):  out[t] = sum_{d=0..16} w[d]    * x[t+d]
// Writes f32 result and bf16 copy (GEMM A input).
// ---------------------------------------------------------------------------
__global__ __launch_bounds__(256) void band_conv_kernel(
    const float* __restrict__ Xf, const float* __restrict__ Xb,
    const float* __restrict__ wf, const float* __restrict__ wb,
    float* __restrict__ Pf, float* __restrict__ Pb,
    __bf16* __restrict__ Af, __bf16* __restrict__ Ab)
{
    const int t   = blockIdx.x;
    const int b   = blockIdx.y;
    const int dir = blockIdx.z;

    const float* X = dir ? Xb : Xf;
    const float* w = dir ? wb : wf;
    float*  P = dir ? Pb : Pf;
    __bf16* A = dir ? Ab : Af;

    float wv[17];
    #pragma unroll
    for (int d = 0; d < 17; ++d) wv[d] = w[d];

    const size_t rowBase = ((size_t)b * SS + t) * DD;
    for (int c = threadIdx.x; c < DD; c += 256) {
        float acc = 0.0f;
        if (dir == 0) {
            #pragma unroll
            for (int d = 0; d <= 16; ++d) {
                const int tt = t - d;
                if (tt >= 0)
                    acc = fmaf(wv[16 - d], X[((size_t)b * SS + tt) * DD + c], acc);
            }
        } else {
            #pragma unroll
            for (int d = 0; d <= 16; ++d) {
                const int tt = t + d;
                if (tt < SS)
                    acc = fmaf(wv[d], X[((size_t)b * SS + tt) * DD + c], acc);
            }
        }
        P[rowBase + c] = acc;
        A[rowBase + c] = (__bf16)acc;
    }
}

// ---------------------------------------------------------------------------
// Fused highway step:  proj = A @ W + b ; [nl | gate] = split(proj)
//   out = sigmoid(gate) * x + (1 - sigmoid(gate)) * relu(nl)
// A:  [M][512] bf16 (== bf16(x)),  X: [M][512] f32,  Wt: [1024][512] bf16 (n-major)
// Block = 256 threads = 8 waves (2 M x 4 N). Wave tile: 64 rows x 32 cols, with
// BOTH the nl columns (n) and gate columns (n+512) accumulated -> 16x v8f accums.
// Epilogue optionally writes Yf (f32), Yb (bf16), and the concat slice of d_out;
// any of the three may be null so dead streams are skipped.
// ---------------------------------------------------------------------------
__global__ __launch_bounds__(256) void highway_gemm_kernel(
    const __bf16* __restrict__ A, const float* __restrict__ X,
    const __bf16* __restrict__ Wt, const float* __restrict__ bias,
    float* __restrict__ Yf, __bf16* __restrict__ Yb,
    float* __restrict__ Out /* stride 1024, dir offset pre-applied */)
{
    const int tid    = threadIdx.x;
    const int lane   = tid & 31;
    const int wave   = tid >> 5;
    const int wm     = wave >> 2;    // 0..1
    const int wn     = wave & 3;     // 0..3
    const int laneLo = lane & 15;
    const int g      = lane >> 4;    // lane half

    const int rowBase = blockIdx.x * 128 + wm * 64;   // M
    const int colBase = blockIdx.y * 128 + wn * 32;   // in [0,512)

    v8f acc_n[4][2];
    v8f acc_g[4][2];
    #pragma unroll
    for (int mi = 0; mi < 4; ++mi)
        #pragma unroll
        for (int ni = 0; ni < 2; ++ni) { acc_n[mi][ni] = (v8f)0.0f; acc_g[mi][ni] = (v8f)0.0f; }

    for (int k0 = 0; k0 < DD; k0 += 32) {
        // A fragments: lane holds row m; 16 contiguous bf16 at K = k0+g*16
        v16bf afrag[4];
        #pragma unroll
        for (int mi = 0; mi < 4; ++mi) {
            const int m = rowBase + mi * 16 + laneLo;
            const __bf16* p = A + (size_t)m * DD + k0 + g * 16;
            v8bf lo = *(const v8bf*)p;
            v8bf hi = *(const v8bf*)(p + 8);
            afrag[mi] = __builtin_shufflevector(lo, hi,
                0, 1, 2, 3, 4, 5, 6, 7, 8, 9, 10, 11, 12, 13, 14, 15);
        }
        // B fragments: lane holds column n; 16 contiguous bf16 at K = k0+g*16
        v16bf bn[2], bg[2];
        #pragma unroll
        for (int ni = 0; ni < 2; ++ni) {
            const int n = colBase + ni * 16 + laneLo;
            bn[ni] = *(const v16bf*)(Wt + (size_t)n * DD + k0 + g * 16);
            bg[ni] = *(const v16bf*)(Wt + (size_t)(n + DD) * DD + k0 + g * 16);
        }
        #pragma unroll
        for (int mi = 0; mi < 4; ++mi)
            #pragma unroll
            for (int ni = 0; ni < 2; ++ni) {
                acc_n[mi][ni] = __builtin_amdgcn_wmma_f32_16x16x32_bf16(
                    false, afrag[mi], false, bn[ni], (short)0, acc_n[mi][ni], false, false);
                acc_g[mi][ni] = __builtin_amdgcn_wmma_f32_16x16x32_bf16(
                    false, afrag[mi], false, bg[ni], (short)0, acc_g[mi][ni], false, false);
            }
    }

    // Epilogue: C/D layout -> VGPR v: m = base + v + 8*g, n = base + laneLo
    #pragma unroll
    for (int mi = 0; mi < 4; ++mi) {
        #pragma unroll
        for (int ni = 0; ni < 2; ++ni) {
            const int n = colBase + ni * 16 + laneLo;
            const float bnl = bias[n];
            const float bgt = bias[n + DD];
            #pragma unroll
            for (int v = 0; v < 8; ++v) {
                const int m = rowBase + mi * 16 + v + g * 8;
                float nl = acc_n[mi][ni][v] + bnl;
                nl = fmaxf(nl, 0.0f);
                const float gt = acc_g[mi][ni][v] + bgt;
                const float sg = 1.0f / (1.0f + __expf(-gt));
                const float xv = X[(size_t)m * DD + n];
                const float o  = sg * xv + (1.0f - sg) * nl;
                if (Yf)  Yf[(size_t)m * DD + n] = o;
                if (Yb)  Yb[(size_t)m * DD + n] = (__bf16)o;
                if (Out) Out[(size_t)m * NPROJ + n] = o;
            }
        }
    }
}

// ---------------------------------------------------------------------------
// Host orchestration
// ---------------------------------------------------------------------------
extern "C" void kernel_launch(void* const* d_in, const int* in_sizes, int n_in,
                              void* d_out, int out_size, void* d_ws, size_t ws_size,
                              hipStream_t stream) {
    const float* inputs  = (const float*)d_in[0];
    // d_in[1] = masks (all ones; unused by the reference computation)
    const float* fw_band = (const float*)d_in[2];
    const float* bw_band = (const float*)d_in[3];
    const float* fw_W    = (const float*)d_in[4];
    const float* fw_b    = (const float*)d_in[5];
    const float* bw_W    = (const float*)d_in[6];
    const float* bw_b    = (const float*)d_in[7];
    float* out = (float*)d_out;

    char* ws = (char*)d_ws;
    const size_t F = (size_t)MM * DD * sizeof(float);   // 64 MB
    const size_t H = (size_t)MM * DD * sizeof(__bf16);  // 32 MB
    float*  Pf  = (float*)(ws + 0 * F);
    float*  Pb  = (float*)(ws + 1 * F);
    float*  Qf  = (float*)(ws + 2 * F);
    float*  Qb  = (float*)(ws + 3 * F);
    float*  Rf  = (float*)(ws + 4 * F);
    float*  Rb  = (float*)(ws + 5 * F);
    __bf16* A0f = (__bf16*)(ws + 6 * F + 0 * H);
    __bf16* A0b = (__bf16*)(ws + 6 * F + 1 * H);
    __bf16* A1f = (__bf16*)(ws + 6 * F + 2 * H);
    __bf16* A1b = (__bf16*)(ws + 6 * F + 3 * H);
    __bf16* Wt  = (__bf16*)(ws + 6 * F + 4 * H);        // 8 matrices x [1024][512]

    // Convert + transpose all 8 weight matrices once.
    wt_transpose_kernel<<<dim3(NPROJ / 32, DD / 32, 8), 256, 0, stream>>>(fw_W, bw_W, Wt);

    const size_t WtM = (size_t)NPROJ * DD;  // elements per matrix
    const dim3 gGrid(MM / 128, DD / 128);   // 256 x 4

    for (int l = 0; l < 2; ++l) {
        const float* inF = (l == 0) ? inputs : Rf;
        const float* inB = (l == 0) ? inputs : Rb;

        band_conv_kernel<<<dim3(SS, BB, 2), 256, 0, stream>>>(
            inF, inB, fw_band + l * 17, bw_band + l * 17, Pf, Pb, A0f, A0b);

        // highway step 0: f32 + bf16 outputs feed step 1; no concat write
        highway_gemm_kernel<<<gGrid, 256, 0, stream>>>(
            A0f, Pf, Wt + (size_t)(0 * 4 + l * 2 + 0) * WtM, fw_b + (l * 2 + 0) * NPROJ,
            Qf, A1f, nullptr);
        highway_gemm_kernel<<<gGrid, 256, 0, stream>>>(
            A0b, Pb, Wt + (size_t)(1 * 4 + l * 2 + 0) * WtM, bw_b + (l * 2 + 0) * NPROJ,
            Qb, A1b, nullptr);

        // highway step 1: writes concat slice of d_out (fwd at +0, bwd at +512).
        // f32 output only needed while a next layer exists; bf16 copy never needed
        // (the next band conv regenerates it) -> skip dead streams.
        float* outF = out + (size_t)l * MM * NPROJ;
        float* outB = outF + DD;
        float* nxtF = (l + 1 < 2) ? Rf : nullptr;
        float* nxtB = (l + 1 < 2) ? Rb : nullptr;
        highway_gemm_kernel<<<gGrid, 256, 0, stream>>>(
            A1f, Qf, Wt + (size_t)(0 * 4 + l * 2 + 1) * WtM, fw_b + (l * 2 + 1) * NPROJ,
            nxtF, nullptr, outF);
        highway_gemm_kernel<<<gGrid, 256, 0, stream>>>(
            A1b, Qb, Wt + (size_t)(1 * 4 + l * 2 + 1) * WtM, bw_b + (l * 2 + 1) * NPROJ,
            nxtB, nullptr, outB);
    }

    (void)in_sizes; (void)n_in; (void)out_size; (void)ws_size;
}
